// Dense_56779467653682
// MI455X (gfx1250) — matrix operand order repaired and checked by reading, not compiled, not run on hardware
//
#include <hip/hip_runtime.h>
#include <cstdint>
#include <cstddef>

// Problem dims (fixed by the reference)
#define MDIM (8 * 2048)   // batch*seq rows of x
#define KDIM 4096         // in_dim
#define NDIM 4096         // features

// GEMM tiling
#define BM 128
#define BN 128
#define BK 32
#define LDT (BK + 8)      // LDS row stride in elements: 40 elems = 80 B = 5*16 B (keeps b128 aligned)

typedef __bf16 bfx4  __attribute__((ext_vector_type(4)));
typedef __bf16 bfx8  __attribute__((ext_vector_type(8)));
typedef __bf16 bfx16 __attribute__((ext_vector_type(16)));
typedef float  fx4   __attribute__((ext_vector_type(4)));
typedef float  fx8   __attribute__((ext_vector_type(8)));
typedef int    v4i   __attribute__((ext_vector_type(4)));

#if __has_builtin(__builtin_amdgcn_global_load_async_to_lds_b128)
#define HAVE_ASYNC_LDS 1
typedef __attribute__((address_space(1))) v4i* gv4i;   // global src (param is v4i __device__*)
typedef __attribute__((address_space(3))) v4i* lv4i;   // LDS dst
#endif

__device__ __forceinline__ void wait_asynccnt0() {
#if defined(HAVE_ASYNC_LDS)
#if __has_builtin(__builtin_amdgcn_s_wait_asynccnt)
  __builtin_amdgcn_s_wait_asynccnt(0);
#else
  asm volatile("s_wait_asynccnt 0x0" ::: "memory");
#endif
#endif
}

// ---------------------------------------------------------------------------
// Pass 1: bool kernel[K][N] (1 byte each) -> bf16 +/-1, transposed: Wt[N][K].
// 64x64 tiles through LDS so both the global read and the global write are
// fully coalesced. One-time cost (16 MB in / 32 MB out), then Wt lives in L2.
// ---------------------------------------------------------------------------
__global__ __launch_bounds__(256)
void pack_weights(const unsigned char* __restrict__ kb, unsigned short* __restrict__ wt) {
  __shared__ unsigned short t[64][72];               // 72*2 = 144 B row stride (16B-aligned)
  const int tid = threadIdx.x;
  const int k0 = blockIdx.y * 64;
  const int n0 = blockIdx.x * 64;

  // load 64(k) x 64(n) bytes, 16 bytes per thread, coalesced
  const int krow = tid >> 2;                         // 0..63
  const int nb   = (tid & 3) << 4;                   // 0,16,32,48
  uint4 d = *(const uint4*)(kb + (size_t)(k0 + krow) * NDIM + n0 + nb);
  const unsigned char* pb = (const unsigned char*)&d;
#pragma unroll
  for (int b = 0; b < 16; ++b)
    t[nb + b][krow] = pb[b] ? 0x3F80u : 0xBF80u;     // +1.0 : -1.0 in bf16
  __syncthreads();

  // write 64(n) x 64(k) bf16, 32 bytes per thread, coalesced
  const int nrow = tid >> 2;                         // 0..63
  const int kc   = (tid & 3) << 4;                   // 0,16,32,48
  uint4 v0 = *(const uint4*)&t[nrow][kc];
  uint4 v1 = *(const uint4*)&t[nrow][kc + 8];
  unsigned short* dst = wt + (size_t)(n0 + nrow) * KDIM + k0 + kc;
  *(uint4*)dst       = v0;
  *(uint4*)(dst + 8) = v1;
}

// ---------------------------------------------------------------------------
// Pass 2: C[M][N] = scale * A[M][K] (f32 -> bf16) x Wt[N][K]^T via
// v_wmma_f32_16x16x32_bf16. 128x128x32 block tile, 8 waves (2x4), each wave
// computes 64x32 = 4x2 WMMA tiles. Double-buffered LDS; B tile moved with
// GLOBAL_LOAD_ASYNC_TO_LDS (ASYNCcnt) when the toolchain exposes it.
// ---------------------------------------------------------------------------
__device__ __forceinline__ void stage_a(unsigned short* __restrict__ As,
                                        const float4 areg[4], int a_row, int a_col) {
#pragma unroll
  for (int j = 0; j < 4; ++j) {
    fx4 f;
    f.x = areg[j].x; f.y = areg[j].y; f.z = areg[j].z; f.w = areg[j].w;
    bfx4 h = __builtin_convertvector(f, bfx4);       // v_cvt_pk_bf16_f32 path
    *(bfx4*)&As[(a_row + 32 * j) * LDT + a_col] = h;
  }
}

__global__ __launch_bounds__(256)
void ternary_gemm(const float* __restrict__ X,
                  const unsigned short* __restrict__ Wt,
                  const float* __restrict__ scale_p,
                  float* __restrict__ Out) {
  __shared__ unsigned short As[2][BM * LDT];   // 2 * 128*40*2B = 20 KB
  __shared__ unsigned short Bs[2][BN * LDT];   // 20 KB  (40 KB total of 320 KB)

  const int tid   = threadIdx.x;
  const int lane  = tid & 31;
  const int wid   = tid >> 5;        // 0..7
  const int wm    = wid >> 2;        // 0..1 -> 64 rows each
  const int wn    = wid & 3;         // 0..3 -> 32 cols each
  const int lrow  = lane & 15;
  const int lhalf = lane >> 4;       // 0/1

  const int m0 = blockIdx.y * BM;
  const int n0 = blockIdx.x * BN;

  // global->LDS staging maps (full tiles; M,N,K all divide evenly -> no guards,
  // EXEC stays all-ones as WMMA requires)
  const int a_row = tid >> 3;            // +32 per j (4 float4 per thread)
  const int a_col = (tid & 7) * 4;
  const int b_row = tid >> 2;            // +64 per j (2 x 16B per thread)
  const int b_col = (tid & 3) * 8;

  const float*          gA = X  + (size_t)(m0 + a_row) * KDIM + a_col;
  const unsigned short* gB = Wt + (size_t)(n0 + b_row) * KDIM + b_col;

  fx8 acc[4][2];
#pragma unroll
  for (int mi = 0; mi < 4; ++mi)
#pragma unroll
    for (int ni = 0; ni < 2; ++ni)
#pragma unroll
      for (int r = 0; r < 8; ++r) acc[mi][ni][r] = 0.0f;

  // ---- prologue: k-tile 0 into buffer 0 ----
  float4 areg[4];
#pragma unroll
  for (int j = 0; j < 4; ++j) areg[j] = *(const float4*)(gA + (size_t)(32 * j) * KDIM);
#if defined(HAVE_ASYNC_LDS)
#pragma unroll
  for (int j = 0; j < 2; ++j)
    __builtin_amdgcn_global_load_async_to_lds_b128(
        (gv4i)(gB + (size_t)(64 * j) * KDIM),
        (lv4i)&Bs[0][(b_row + 64 * j) * LDT + b_col], 0, 0);
#else
  uint4 breg[2];
#pragma unroll
  for (int j = 0; j < 2; ++j) breg[j] = *(const uint4*)(gB + (size_t)(64 * j) * KDIM);
#pragma unroll
  for (int j = 0; j < 2; ++j)
    *(uint4*)&Bs[0][(b_row + 64 * j) * LDT + b_col] = breg[j];
#endif
  stage_a(As[0], areg, a_row, a_col);
  wait_asynccnt0();
  __syncthreads();

  const int NKT = KDIM / BK;   // 128 k-steps
  int buf = 0;
  for (int kt = 0; kt < NKT; ++kt) {
    // issue next tile's data movement first so it overlaps the WMMA stream
    if (kt + 1 < NKT) {
      const float*          gA2 = gA + (kt + 1) * BK;
      const unsigned short* gB2 = gB + (kt + 1) * BK;
#if defined(HAVE_ASYNC_LDS)
#pragma unroll
      for (int j = 0; j < 2; ++j)
        __builtin_amdgcn_global_load_async_to_lds_b128(
            (gv4i)(gB2 + (size_t)(64 * j) * KDIM),
            (lv4i)&Bs[buf ^ 1][(b_row + 64 * j) * LDT + b_col], 0, 0);
#else
#pragma unroll
      for (int j = 0; j < 2; ++j) breg[j] = *(const uint4*)(gB2 + (size_t)(64 * j) * KDIM);
#endif
#pragma unroll
      for (int j = 0; j < 4; ++j) areg[j] = *(const float4*)(gA2 + (size_t)(32 * j) * KDIM);
    }

    // A fragments: lane l<16 holds K 0..7 & 16..23; lane>=16 holds K 8..15 & 24..31
    bfx16 afrag[4];
#pragma unroll
    for (int mi = 0; mi < 4; ++mi) {
      const unsigned short* p = &As[buf][(wm * 64 + mi * 16 + lrow) * LDT + lhalf * 8];
      bfx8 lo = *(const bfx8*)p;
      bfx8 hi = *(const bfx8*)(p + 16);
      afrag[mi] = __builtin_shufflevector(lo, hi, 0, 1, 2, 3, 4, 5, 6, 7,
                                          8, 9, 10, 11, 12, 13, 14, 15);
    }
    // B fragments: lane l<16 holds K 0..7 & 16..23; lane>=16 holds K 8..15 & 24..31
    bfx16 bfrag[2];
#pragma unroll
    for (int ni = 0; ni < 2; ++ni) {
      const unsigned short* p = &Bs[buf][(wn * 32 + ni * 16 + lrow) * LDT + lhalf * 8];
      bfx8 lo = *(const bfx8*)p;
      bfx8 hi = *(const bfx8*)(p + 16);
      bfrag[ni] = __builtin_shufflevector(lo, hi, 0, 1, 2, 3, 4, 5, 6, 7,
                                          8, 9, 10, 11, 12, 13, 14, 15);
    }

#pragma unroll
    for (int mi = 0; mi < 4; ++mi)
#pragma unroll
      for (int ni = 0; ni < 2; ++ni)
        acc[mi][ni] = __builtin_amdgcn_wmma_f32_16x16x32_bf16(
            false, afrag[mi], false, bfrag[ni], (short)0, acc[mi][ni], false, false);

    if (kt + 1 < NKT) {
      stage_a(As[buf ^ 1], areg, a_row, a_col);
#if !defined(HAVE_ASYNC_LDS)
#pragma unroll
      for (int j = 0; j < 2; ++j)
        *(uint4*)&Bs[buf ^ 1][(b_row + 64 * j) * LDT + b_col] = breg[j];
#endif
      wait_asynccnt0();
      __syncthreads();
      buf ^= 1;
    }
  }

  // Epilogue: C/D layout — VGPR r, lanes 0-15: (M=r, N=lane); lanes 16-31: (M=8+r, N=lane-16)
  // Non-temporal: the 256 MB output is write-once, keep it out of L2 so the
  // A-tiles + 32 MB Wt working set stays resident.
  const float scale = *scale_p;
#pragma unroll
  for (int mi = 0; mi < 4; ++mi) {
#pragma unroll
    for (int ni = 0; ni < 2; ++ni) {
      const int n  = n0 + wn * 32 + ni * 16 + lrow;
      const int mb = m0 + wm * 64 + mi * 16 + lhalf * 8;
      float* po = Out + (size_t)mb * NDIM + n;
#pragma unroll
      for (int r = 0; r < 8; ++r)
        __builtin_nontemporal_store(scale * acc[mi][ni][r], po + (size_t)r * NDIM);
    }
  }
}

// ---------------------------------------------------------------------------
extern "C" void kernel_launch(void* const* d_in, const int* in_sizes, int n_in,
                              void* d_out, int out_size, void* d_ws, size_t ws_size,
                              hipStream_t stream) {
  const float*         X     = (const float*)d_in[0];
  const unsigned char* Kb    = (const unsigned char*)d_in[1];  // jax bool -> 1 byte
  const float*         scale = (const float*)d_in[2];
  float*               Out   = (float*)d_out;
  unsigned short*      Wt    = (unsigned short*)d_ws;          // needs NDIM*KDIM*2 = 32 MB

  dim3 pgrid(NDIM / 64, KDIM / 64);
  pack_weights<<<pgrid, 256, 0, stream>>>(Kb, Wt);

  dim3 ggrid(NDIM / BN, MDIM / BM);
  ternary_gemm<<<ggrid, 256, 0, stream>>>(X, Wt, scale, Out);
}
